// multi_modal_layer_28063316312631
// MI455X (gfx1250) — compile-verified, matrix operand order checked by reading, not run on hardware
//
#include <hip/hip_runtime.h>
#include <hip/hip_bf16.h>
#include <cstdint>
#include <cstddef>

// Problem constants (from reference)
#define B_   16
#define S_   80
#define L_   200
#define LP_  224      // L padded to multiple of 32 for WMMA K-dim
#define M_   1024
#define T_   1024
#define E_   512
#define H_   512
#define V_   10000
#define CIN_ 2560     // E+T+M == H+T+M
#define KC_  512      // K-chunk staged in LDS (16x512 bf16 = 16KB)

typedef __attribute__((ext_vector_type(16))) __bf16 v16bf;
typedef __attribute__((ext_vector_type(8)))  float  v8f;

union FragBF { uint4 q[2]; v16bf v; };

__device__ __forceinline__ float sigf(float x) { return 1.0f / (1.0f + expf(-x)); }

// ---------------------------------------------------------------------------
// CDNA5 async global->LDS copy (16 bytes per lane), tracked by ASYNCcnt.
// GV addressing: vdst = LDS byte address, vaddr = 64-bit global address.
// ---------------------------------------------------------------------------
__device__ __forceinline__ void async_copy16(const __bf16* g, __bf16* l) {
    unsigned lds = (unsigned)(unsigned long)(__attribute__((address_space(3))) __bf16*)l;
    asm volatile("global_load_async_to_lds_b128 %0, %1, off"
                 :: "v"(lds), "v"((unsigned long)g) : "memory");
}
__device__ __forceinline__ void wait_async0() {
    asm volatile("s_wait_asynccnt 0x0" ::: "memory");
}

// ---------------------------------------------------------------------------
// Elementwise / conversion kernels
// ---------------------------------------------------------------------------
__global__ void f2bf(const float* __restrict__ s, __bf16* __restrict__ d, long n) {
    for (long i = blockIdx.x * (long)blockDim.x + threadIdx.x; i < n;
         i += (long)gridDim.x * blockDim.x)
        d[i] = (__bf16)s[i];
}

// (B,L,W) f32 -> (B,LP,W) bf16, zero-padding rows L..LP-1
__global__ void f2bf_padrows(const float* __restrict__ s, __bf16* __restrict__ d,
                             int rowsIn, int rowsOut, int width, long n) {
    for (long i = blockIdx.x * (long)blockDim.x + threadIdx.x; i < n;
         i += (long)gridDim.x * blockDim.x) {
        int  c = (int)(i % width);
        long r = i / width;
        int  l = (int)(r % rowsOut);
        long b = r / rowsOut;
        d[i] = (l < rowsIn) ? (__bf16)s[(b * rowsIn + l) * (long)width + c] : (__bf16)0.0f;
    }
}

// dense (rows,width) f32 -> bf16 strided columns of a wider matrix
__global__ void f2bf_cols(const float* __restrict__ s, int width,
                          __bf16* __restrict__ d, long ldd, long colOff, long n) {
    for (long i = blockIdx.x * (long)blockDim.x + threadIdx.x; i < n;
         i += (long)gridDim.x * blockDim.x) {
        long r = i / width; int c = (int)(i % width);
        d[r * ldd + colOff + c] = (__bf16)s[i];
    }
}

__global__ void pack_wcat(const float* __restrict__ Wih, const float* __restrict__ Whh,
                          __bf16* __restrict__ d, long n) {  // (2048,1024)
    for (long i = blockIdx.x * (long)blockDim.x + threadIdx.x; i < n;
         i += (long)gridDim.x * blockDim.x) {
        long r = i >> 10; int c = (int)(i & 1023);
        d[i] = (__bf16)((c < H_) ? Wih[r * H_ + c] : Whh[r * H_ + (c - H_)]);
    }
}

__global__ void add_bias2(const float* a, const float* b, float* d, long n) {
    for (long i = blockIdx.x * (long)blockDim.x + threadIdx.x; i < n;
         i += (long)gridDim.x * blockDim.x) d[i] = a[i] + b[i];
}

__global__ void zero_f32(float* p, long n) {
    for (long i = blockIdx.x * (long)blockDim.x + threadIdx.x; i < n;
         i += (long)gridDim.x * blockDim.x) p[i] = 0.0f;
}
__global__ void zero_bf16(__bf16* p, long n) {
    for (long i = blockIdx.x * (long)blockDim.x + threadIdx.x; i < n;
         i += (long)gridDim.x * blockDim.x) p[i] = (__bf16)0.0f;
}

// ---------------------------------------------------------------------------
// WMMA GEMM (register path): C = A * W^T + bias. One 16x16 tile per wave.
// Used for the latency-critical LSTM steps (M=16, no barriers on the
// recurrence critical path).
// ---------------------------------------------------------------------------
__global__ void __launch_bounds__(128)
gemm_nt_bf16(const __bf16* __restrict__ A, long lda,
             const __bf16* __restrict__ W, long ldb,
             const float* __restrict__ bias,
             float* __restrict__ C, long ldc,
             int Ntiles, int Kdim) {
    int wave = threadIdx.x >> 5;
    int lane = threadIdx.x & 31;
    int nt = blockIdx.x * 4 + wave;
    int mt = blockIdx.y;
    if (nt >= Ntiles) return;
    int half = lane >> 4;
    int r16  = lane & 15;
    const __bf16* ap = A + (long)(mt * 16 + r16) * lda + half * 8;
    const __bf16* bp = W + (long)(nt * 16 + r16) * ldb + half * 8;
    int col = nt * 16 + r16;
    float bv = bias ? bias[col] : 0.0f;
    v8f acc;
#pragma unroll
    for (int i = 0; i < 8; ++i) acc[i] = bv;
    for (int k = 0; k < Kdim; k += 32) {
        FragBF a, b;
        a.q[0] = *(const uint4*)(ap + k);
        a.q[1] = *(const uint4*)(ap + k + 16);
        b.q[0] = *(const uint4*)(bp + k);
        b.q[1] = *(const uint4*)(bp + k + 16);
        acc = __builtin_amdgcn_wmma_f32_16x16x32_bf16(false, a.v, false, b.v,
                                                      (short)0, acc, false, false);
    }
    float* cp = C + (long)(mt * 16 + half * 8) * ldc + col;
#pragma unroll
    for (int i = 0; i < 8; ++i) cp[(long)i * ldc] = acc[i];
}

// ---------------------------------------------------------------------------
// WMMA GEMM (async-LDS path): the 16-row A panel for this block is staged in
// LDS in KC_-wide chunks with global_load_async_to_lds_b128 (ASYNCcnt) and
// shared by all 4 waves; B streams from global/L2. Used for the large GEMMs.
// Kdim must be a multiple of KC_. Inactive tail waves (N=625 case) keep
// participating in copies/barriers and only skip the WMMA compute, so EXEC
// is all-ones for every wmma.
// ---------------------------------------------------------------------------
__global__ void __launch_bounds__(128)
gemm_nt_bf16_lds(const __bf16* __restrict__ A, long lda,
                 const __bf16* __restrict__ W, long ldb,
                 const float* __restrict__ bias,
                 float* __restrict__ C, long ldc,
                 int Ntiles, int Kdim) {
    __shared__ __bf16 As[16 * KC_];
    int tid  = threadIdx.x;
    int wave = tid >> 5;
    int lane = tid & 31;
    int nt = blockIdx.x * 4 + wave;
    int mt = blockIdx.y;
    bool active = (nt < Ntiles);
    int ntc = active ? nt : 0;
    int half = lane >> 4;
    int r16  = lane & 15;
    const __bf16* bp = W + (long)(ntc * 16 + r16) * ldb + half * 8;
    int col = ntc * 16 + r16;
    float bv = (bias && active) ? bias[col] : 0.0f;
    v8f acc;
#pragma unroll
    for (int i = 0; i < 8; ++i) acc[i] = bv;

    for (int k0 = 0; k0 < Kdim; k0 += KC_) {
        __syncthreads();   // previous chunk fully consumed before overwrite
        // cooperative async stage of A[mt*16..+16) x [k0, k0+KC_):
        // 16*KC_/8 = 1024 16-byte chunks over 128 threads -> 8 each
#pragma unroll
        for (int c = tid; c < 16 * KC_ / 8; c += 128) {
            int row = c >> 6;          // KC_*2/16 = 64 chunks per row
            int cc  = c & 63;
            async_copy16(A + (long)(mt * 16 + row) * lda + k0 + cc * 8,
                         &As[row * KC_ + cc * 8]);
        }
        wait_async0();
        __syncthreads();
        if (active) {
            const __bf16* ap = &As[r16 * KC_ + half * 8];
            for (int k = 0; k < KC_; k += 32) {
                FragBF a, b;
                a.q[0] = *(const uint4*)(ap + k);        // ds_load_b128
                a.q[1] = *(const uint4*)(ap + k + 16);
                b.q[0] = *(const uint4*)(bp + k0 + k);   // global_load_b128
                b.q[1] = *(const uint4*)(bp + k0 + k + 16);
                acc = __builtin_amdgcn_wmma_f32_16x16x32_bf16(false, a.v, false, b.v,
                                                              (short)0, acc, false, false);
            }
        }
    }
    if (active) {
        float* cp = C + (long)(mt * 16 + half * 8) * ldc + col;
#pragma unroll
        for (int i = 0; i < 8; ++i) cp[(long)i * ldc] = acc[i];
    }
}

// ---------------------------------------------------------------------------
// WMMA batched GEMM (NN): C = A(SxK) * Bm(KxN) per batch; bf16 output into
// the final-concat buffer. B-fragment gathered per element (coalesced across
// lanes for fixed k).
// ---------------------------------------------------------------------------
__global__ void __launch_bounds__(128)
gemm_nn_bf16(const __bf16* __restrict__ Abase, long aBatchStride, long lda,
             const __bf16* __restrict__ Bbase, long bBatchStride, long ldb,
             __bf16* __restrict__ Cbase, long cRowStride, long colOff,
             int Ntiles, int Kdim) {
    int wave = threadIdx.x >> 5;
    int lane = threadIdx.x & 31;
    int nt = blockIdx.x * 4 + wave;
    int mt = blockIdx.y;
    int bb = blockIdx.z;
    if (nt >= Ntiles) return;
    int half = lane >> 4;
    int r16  = lane & 15;
    const __bf16* ap   = Abase + (long)bb * aBatchStride + (long)(mt * 16 + r16) * lda + half * 8;
    const __bf16* bcol = Bbase + (long)bb * bBatchStride + (nt * 16 + r16);
    v8f acc;
#pragma unroll
    for (int i = 0; i < 8; ++i) acc[i] = 0.0f;
    for (int k = 0; k < Kdim; k += 32) {
        FragBF a, b;
        a.q[0] = *(const uint4*)(ap + k);
        a.q[1] = *(const uint4*)(ap + k + 16);
#pragma unroll
        for (int e = 0; e < 16; ++e) {
            int kk = k + ((e >= 8) ? 16 : 0) + half * 8 + (e & 7);
            b.v[e] = bcol[(long)kk * ldb];
        }
        acc = __builtin_amdgcn_wmma_f32_16x16x32_bf16(false, a.v, false, b.v,
                                                      (short)0, acc, false, false);
    }
#pragma unroll
    for (int i = 0; i < 8; ++i) {
        long row = (long)bb * S_ + mt * 16 + half * 8 + i;
        Cbase[row * cRowStride + colOff + nt * 16 + r16] = (__bf16)acc[i];
    }
}

// ---------------------------------------------------------------------------
// y = g * softmax(g) over last dim, written as bf16 into a column slice
// ---------------------------------------------------------------------------
__global__ void __launch_bounds__(256)
mul_softmax_row(const float* __restrict__ G, int width,
                __bf16* __restrict__ dst, long ldd, long colOff) {
    long row = blockIdx.x;
    const float* g = G + row * (long)width;
    __shared__ float red[256];
    float m = -1e30f;
    for (int i = threadIdx.x; i < width; i += 256) m = fmaxf(m, g[i]);
    red[threadIdx.x] = m; __syncthreads();
    for (int s = 128; s > 0; s >>= 1) {
        if (threadIdx.x < s) red[threadIdx.x] = fmaxf(red[threadIdx.x], red[threadIdx.x + s]);
        __syncthreads();
    }
    m = red[0]; __syncthreads();
    float sum = 0.0f;
    for (int i = threadIdx.x; i < width; i += 256) sum += expf(g[i] - m);
    red[threadIdx.x] = sum; __syncthreads();
    for (int s = 128; s > 0; s >>= 1) {
        if (threadIdx.x < s) red[threadIdx.x] += red[threadIdx.x + s];
        __syncthreads();
    }
    float inv = 1.0f / red[0];
    for (int i = threadIdx.x; i < width; i += 256) {
        float v = g[i];
        dst[row * ldd + colOff + i] = (__bf16)(v * expf(v - m) * inv);
    }
}

// avgpool2d(tempo,10) ++ avgpool2d(motion,10) -> (B, 8, 204) f32
__global__ void avgpool_concat(const float* __restrict__ tempo,
                               const float* __restrict__ motion,
                               float* __restrict__ out) {
    int idx = blockIdx.x * blockDim.x + threadIdx.x;
    if (idx >= B_ * 8 * 204) return;
    int c  = idx % 204;
    int sr = (idx / 204) % 8;
    int b  = idx / (204 * 8);
    const float* src; int ch;
    if (c < 102) { src = tempo;  ch = c * 10; }
    else         { src = motion; ch = (c - 102) * 10; }
    float s = 0.0f;
    for (int ds = 0; ds < 10; ++ds)
        for (int dc = 0; dc < 10; ++dc)
            s += src[((long)b * S_ + sr * 10 + ds) * 1024 + ch + dc];
    out[idx] = s * 0.01f;
}

// m0 = mean_feat(128x204) @ Wli^T(512x204) + bli  (tiny, plain f32)
__global__ void m0_gemm(const float* __restrict__ mf, const float* __restrict__ Wli,
                        const float* __restrict__ bli, float* __restrict__ m0) {
    int idx = blockIdx.x * blockDim.x + threadIdx.x;
    if (idx >= 128 * 512) return;
    int row = idx >> 9, n = idx & 511;
    const float* a = mf + (long)row * 204;
    const float* w = Wli + (long)n * 204;
    float s = bli[n];
    for (int k = 0; k < 204; ++k) s += a[k] * w[k];
    m0[idx] = s;
}

// pack [x_t | h] -> (16,1024) bf16 for the fused gate GEMM
__global__ void pack_xh(const __bf16* __restrict__ xbase, long rowStride, long tOff,
                        const __bf16* __restrict__ h, __bf16* __restrict__ xh) {
    int idx = blockIdx.x * blockDim.x + threadIdx.x;
    if (idx >= 16 * 1024) return;
    int r = idx >> 10, c = idx & 1023;
    xh[idx] = (c < H_) ? xbase[(long)r * rowStride + tOff + c] : h[(long)r * H_ + (c - H_)];
}

// LSTM gate nonlinearity; optionally emits act = tanh(h) (f32 + bf16 concat col)
__global__ void lstm_gate(const float* __restrict__ g, float* __restrict__ c,
                          __bf16* __restrict__ h_bf,
                          float* __restrict__ act, __bf16* __restrict__ fin,
                          int t, int writeAct) {
    int idx = blockIdx.x * blockDim.x + threadIdx.x;
    if (idx >= B_ * H_) return;
    int b = idx >> 9, j = idx & 511;
    const float* gr = g + (long)b * (4 * H_);
    float i_ = sigf(gr[j]);
    float f_ = sigf(gr[H_ + j]);
    float gg = tanhf(gr[2 * H_ + j]);
    float o_ = sigf(gr[3 * H_ + j]);
    float cn = f_ * c[idx] + i_ * gg;
    float hn = o_ * tanhf(cn);
    c[idx] = cn;
    h_bf[idx] = (__bf16)hn;
    if (writeAct) {
        float a = tanhf(hn);
        long row = (long)b * S_ + t;
        act[row * H_ + j] = a;
        fin[row * CIN_ + j] = (__bf16)a;
    }
}

// out[row] = dot(X[row,:], w)
__global__ void __launch_bounds__(256)
rowdot(const float* __restrict__ X, int width, const float* __restrict__ w,
       float* __restrict__ out) {
    long row = blockIdx.x;
    const float* x = X + row * (long)width;
    __shared__ float red[256];
    float s = 0.0f;
    for (int i = threadIdx.x; i < width; i += 256) s += x[i] * w[i];
    red[threadIdx.x] = s; __syncthreads();
    for (int st = 128; st > 0; st >>= 1) {
        if (threadIdx.x < st) red[threadIdx.x] += red[threadIdx.x + st];
        __syncthreads();
    }
    if (threadIdx.x == 0) out[row] = red[0];
}

// masked additive-attention softmax weights -> bf16 (rows = B*S, width LP_)
__global__ void __launch_bounds__(256)
attn_weights(const float* __restrict__ U, const float* __restrict__ V,
             const int* __restrict__ lens, const float* __restrict__ bsc,
             __bf16* __restrict__ Wout) {
    int row = blockIdx.x;
    int b = row / S_;
    int len = lens[b];
    float u = U[row] + bsc[0];
    const float* v = V + (long)b * L_;
    __shared__ float red[256];
    float m = -1e30f;
    for (int i = threadIdx.x; i < len; i += 256) m = fmaxf(m, u + v[i]);
    red[threadIdx.x] = m; __syncthreads();
    for (int s = 128; s > 0; s >>= 1) {
        if (threadIdx.x < s) red[threadIdx.x] = fmaxf(red[threadIdx.x], red[threadIdx.x + s]);
        __syncthreads();
    }
    m = red[0]; __syncthreads();
    float sum = 0.0f;
    for (int i = threadIdx.x; i < len; i += 256) sum += expf(u + v[i] - m);
    red[threadIdx.x] = sum; __syncthreads();
    for (int s = 128; s > 0; s >>= 1) {
        if (threadIdx.x < s) red[threadIdx.x] += red[threadIdx.x + s];
        __syncthreads();
    }
    float inv = 1.0f / red[0];
    for (int i = threadIdx.x; i < LP_; i += 256) {
        float wv = (i < len) ? expf(u + v[i] - m) * inv : 0.0f;
        Wout[(long)row * LP_ + i] = (__bf16)wv;
    }
}

// in-place log_softmax along rows of width V_
__global__ void __launch_bounds__(256)
logsoftmax_row(float* __restrict__ X, int width) {
    long row = blockIdx.x;
    float* x = X + row * (long)width;
    __shared__ float red[256];
    float m = -1e30f;
    for (int i = threadIdx.x; i < width; i += 256) m = fmaxf(m, x[i]);
    red[threadIdx.x] = m; __syncthreads();
    for (int s = 128; s > 0; s >>= 1) {
        if (threadIdx.x < s) red[threadIdx.x] = fmaxf(red[threadIdx.x], red[threadIdx.x + s]);
        __syncthreads();
    }
    m = red[0]; __syncthreads();
    float sum = 0.0f;
    for (int i = threadIdx.x; i < width; i += 256) sum += expf(x[i] - m);
    red[threadIdx.x] = sum; __syncthreads();
    for (int s = 128; s > 0; s >>= 1) {
        if (threadIdx.x < s) red[threadIdx.x] += red[threadIdx.x + s];
        __syncthreads();
    }
    float lse = m + logf(red[0]);
    for (int i = threadIdx.x; i < width; i += 256) x[i] -= lse;
}

// ---------------------------------------------------------------------------
extern "C" void kernel_launch(void* const* d_in, const int* in_sizes, int n_in,
                              void* d_out, int out_size, void* d_ws, size_t ws_size,
                              hipStream_t stream) {
    const float* mfeat = (const float*)d_in[0];
    const float* tfeat = (const float*)d_in[1];
    const float* attm  = (const float*)d_in[2];
    const float* attt  = (const float*)d_in[3];
    const float* emb   = (const float*)d_in[4];
    const int*   mlens = (const int*)d_in[5];
    const int*   tlens = (const int*)d_in[6];
    const float* Wm  = (const float*)d_in[8];
    const float* bm  = (const float*)d_in[9];
    const float* Wt  = (const float*)d_in[10];
    const float* bt  = (const float*)d_in[11];
    const float* Wmm = (const float*)d_in[12];
    const float* bmm = (const float*)d_in[13];
    const float* Wli = (const float*)d_in[14];
    const float* bli = (const float*)d_in[15];
    const float* Wih = (const float*)d_in[16];
    const float* Whh = (const float*)d_in[17];
    const float* bih = (const float*)d_in[18];
    const float* bhh = (const float*)d_in[19];
    const float* Wsm = (const float*)d_in[20];
    const float* bsm = (const float*)d_in[21];
    const float* Wst = (const float*)d_in[22];
    const float* bst = (const float*)d_in[23];
    const float* Wo  = (const float*)d_in[24];
    const float* bo  = (const float*)d_in[25];
    float* out = (float*)d_out;

    char* base = (char*)d_ws; size_t off = 0;
    auto alloc = [&](size_t bytes) -> void* {
        void* p = base + off; off = (off + bytes + 255) & ~(size_t)255; return p;
    };
    const long RS = (long)B_ * S_;   // 1280 rows

    __bf16* Wm_bf   = (__bf16*)alloc((size_t)M_ * M_ * 2);
    __bf16* Wt_bf   = (__bf16*)alloc((size_t)T_ * T_ * 2);
    __bf16* Wmm_bf  = (__bf16*)alloc((size_t)H_ * CIN_ * 2);
    __bf16* Wcat_bf = (__bf16*)alloc((size_t)4 * H_ * 1024 * 2);
    __bf16* Wo_bf   = (__bf16*)alloc((size_t)V_ * CIN_ * 2);
    __bf16* attm_bf = (__bf16*)alloc((size_t)RS * M_ * 2);
    __bf16* attt_bf = (__bf16*)alloc((size_t)RS * T_ * 2);
    __bf16* mfeat_bf= (__bf16*)alloc((size_t)B_ * LP_ * M_ * 2);
    __bf16* tfeat_bf= (__bf16*)alloc((size_t)B_ * LP_ * T_ * 2);
    float*  gtmp    = (float*) alloc((size_t)RS * M_ * 4);
    __bf16* mm_inp  = (__bf16*)alloc((size_t)RS * CIN_ * 2);
    float*  mm_out  = (float*) alloc((size_t)RS * H_ * 4);
    __bf16* mm_out_bf=(__bf16*)alloc((size_t)RS * H_ * 2);
    float*  meanf   = (float*) alloc((size_t)B_ * 8 * 204 * 4);
    float*  m0      = (float*) alloc((size_t)B_ * 8 * H_ * 4);
    __bf16* m0_bf   = (__bf16*)alloc((size_t)B_ * 8 * H_ * 2);
    float*  biasL   = (float*) alloc((size_t)4 * H_ * 4);
    __bf16* xh      = (__bf16*)alloc((size_t)B_ * 1024 * 2);
    float*  gates   = (float*) alloc((size_t)B_ * 4 * H_ * 4);
    float*  cstate  = (float*) alloc((size_t)B_ * H_ * 4);
    __bf16* h_bf    = (__bf16*)alloc((size_t)B_ * H_ * 2);
    float*  act     = (float*) alloc((size_t)RS * H_ * 4);
    __bf16* fin     = (__bf16*)alloc((size_t)RS * CIN_ * 2);
    float*  Um      = (float*) alloc((size_t)RS * 4);
    float*  Ut      = (float*) alloc((size_t)RS * 4);
    float*  Vm      = (float*) alloc((size_t)B_ * L_ * 4);
    float*  Vt      = (float*) alloc((size_t)B_ * L_ * 4);
    __bf16* wgt_m   = (__bf16*)alloc((size_t)RS * LP_ * 2);
    __bf16* wgt_t   = (__bf16*)alloc((size_t)RS * LP_ * 2);

    auto blocks = [](long n) { return dim3((unsigned)((n + 255) / 256)); };
    dim3 thr(256);

    // ---- weight / input conversions ----
    f2bf<<<blocks((long)M_ * M_), thr, 0, stream>>>(Wm, Wm_bf, (long)M_ * M_);
    f2bf<<<blocks((long)T_ * T_), thr, 0, stream>>>(Wt, Wt_bf, (long)T_ * T_);
    f2bf<<<blocks((long)H_ * CIN_), thr, 0, stream>>>(Wmm, Wmm_bf, (long)H_ * CIN_);
    pack_wcat<<<blocks((long)4 * H_ * 1024), thr, 0, stream>>>(Wih, Whh, Wcat_bf, (long)4 * H_ * 1024);
    f2bf<<<blocks((long)V_ * CIN_), thr, 0, stream>>>(Wo, Wo_bf, (long)V_ * CIN_);
    f2bf<<<blocks(RS * M_), thr, 0, stream>>>(attm, attm_bf, RS * M_);
    f2bf<<<blocks(RS * T_), thr, 0, stream>>>(attt, attt_bf, RS * T_);
    f2bf_padrows<<<blocks((long)B_ * LP_ * M_), thr, 0, stream>>>(mfeat, mfeat_bf, L_, LP_, M_, (long)B_ * LP_ * M_);
    f2bf_padrows<<<blocks((long)B_ * LP_ * T_), thr, 0, stream>>>(tfeat, tfeat_bf, L_, LP_, T_, (long)B_ * LP_ * T_);
    f2bf_cols<<<blocks(RS * E_), thr, 0, stream>>>(emb, E_, mm_inp, CIN_, 0, RS * E_);
    add_bias2<<<blocks((long)4 * H_), thr, 0, stream>>>(bih, bhh, biasL, (long)4 * H_);

    // ---- stage 1: gated projections (async-LDS WMMA path) ----
    gemm_nt_bf16_lds<<<dim3(M_ / 64, RS / 16), dim3(128), 0, stream>>>(
        attm_bf, M_, Wm_bf, M_, bm, gtmp, M_, M_ / 16, M_);
    mul_softmax_row<<<dim3((unsigned)RS), thr, 0, stream>>>(gtmp, M_, mm_inp, CIN_, E_ + T_);
    gemm_nt_bf16_lds<<<dim3(T_ / 64, RS / 16), dim3(128), 0, stream>>>(
        attt_bf, T_, Wt_bf, T_, bt, gtmp, T_, T_ / 16, T_);
    mul_softmax_row<<<dim3((unsigned)RS), thr, 0, stream>>>(gtmp, T_, mm_inp, CIN_, E_);

    // ---- stage 2: fusion GEMM ----
    gemm_nt_bf16_lds<<<dim3(H_ / 64, RS / 16), dim3(128), 0, stream>>>(
        mm_inp, CIN_, Wmm_bf, CIN_, bmm, mm_out, H_, H_ / 16, CIN_);
    f2bf<<<blocks(RS * H_), thr, 0, stream>>>(mm_out, mm_out_bf, RS * H_);

    // ---- stage 3: pooled warm-up input ----
    avgpool_concat<<<blocks((long)B_ * 8 * 204), thr, 0, stream>>>(attt, attm, meanf);
    m0_gemm<<<blocks((long)128 * 512), thr, 0, stream>>>(meanf, Wli, bli, m0);
    f2bf<<<blocks((long)B_ * 8 * H_), thr, 0, stream>>>(m0, m0_bf, (long)B_ * 8 * H_);

    // ---- LSTMs (register-path GEMM, short critical path) ----
    zero_f32<<<blocks((long)B_ * H_), thr, 0, stream>>>(cstate, (long)B_ * H_);
    zero_bf16<<<blocks((long)B_ * H_), thr, 0, stream>>>(h_bf, (long)B_ * H_);
    for (int t = 0; t < 8; ++t) {   // warm-up LSTM over m0
        pack_xh<<<blocks((long)B_ * 1024), thr, 0, stream>>>(m0_bf, (long)8 * H_, (long)t * H_, h_bf, xh);
        gemm_nt_bf16<<<dim3(128 / 4, 1), dim3(128), 0, stream>>>(
            xh, 1024, Wcat_bf, 1024, biasL, gates, 4 * H_, (4 * H_) / 16, 1024);
        lstm_gate<<<blocks((long)B_ * H_), thr, 0, stream>>>(gates, cstate, h_bf, act, fin, t, 0);
    }
    for (int t = 0; t < S_; ++t) {  // main LSTM over mm_out
        pack_xh<<<blocks((long)B_ * 1024), thr, 0, stream>>>(mm_out_bf, (long)S_ * H_, (long)t * H_, h_bf, xh);
        gemm_nt_bf16<<<dim3(128 / 4, 1), dim3(128), 0, stream>>>(
            xh, 1024, Wcat_bf, 1024, biasL, gates, 4 * H_, (4 * H_) / 16, 1024);
        lstm_gate<<<blocks((long)B_ * H_), thr, 0, stream>>>(gates, cstate, h_bf, act, fin, t, 1);
    }

    // ---- stage 6: additive attentions ----
    rowdot<<<dim3((unsigned)RS), thr, 0, stream>>>(act, H_, Wsm, Um);
    rowdot<<<dim3((unsigned)(B_ * L_)), thr, 0, stream>>>(mfeat, M_, Wsm + H_, Vm);
    rowdot<<<dim3((unsigned)RS), thr, 0, stream>>>(act, H_, Wst, Ut);
    rowdot<<<dim3((unsigned)(B_ * L_)), thr, 0, stream>>>(tfeat, T_, Wst + H_, Vt);
    attn_weights<<<dim3((unsigned)RS), thr, 0, stream>>>(Um, Vm, mlens, bsm, wgt_m);
    attn_weights<<<dim3((unsigned)RS), thr, 0, stream>>>(Ut, Vt, tlens, bst, wgt_t);
    // att_t2 -> fin cols [H, H+T); att_m2 -> fin cols [H+T, H+T+M)
    gemm_nn_bf16<<<dim3(T_ / 64, S_ / 16, B_), dim3(128), 0, stream>>>(
        wgt_t, (long)S_ * LP_, LP_, tfeat_bf, (long)LP_ * T_, T_,
        fin, CIN_, H_, T_ / 16, LP_);
    gemm_nn_bf16<<<dim3(M_ / 64, S_ / 16, B_), dim3(128), 0, stream>>>(
        wgt_m, (long)S_ * LP_, LP_, mfeat_bf, (long)LP_ * M_, M_,
        fin, CIN_, H_ + T_, M_ / 16, LP_);

    // ---- stage 7: output projection (async-LDS WMMA path) + log-softmax ----
    int ntiles = V_ / 16;                         // 625
    gemm_nt_bf16_lds<<<dim3((ntiles + 3) / 4, RS / 16), dim3(128), 0, stream>>>(
        fin, CIN_, Wo_bf, CIN_, bo, out, V_, ntiles, CIN_);
    logsoftmax_row<<<dim3((unsigned)RS), thr, 0, stream>>>(out, V_);
}
